// ExtractorMLP_8993661518313
// MI455X (gfx1250) — matrix operand order repaired and checked by reading, not compiled, not run on hardware
//
#include <hip/hip_runtime.h>
#include <hip/hip_bf16.h>

typedef __attribute__((ext_vector_type(16))) __bf16 bf16x16;
typedef __attribute__((ext_vector_type(8)))  __bf16 bf16x8;
typedef __attribute__((ext_vector_type(8)))  float  f32x8;

#define GSEG 128
#define EPS  1e-5f

__device__ __forceinline__ unsigned short f32_to_bf16(float f) {
    unsigned int u = __float_as_uint(f);
    unsigned int r = u + 0x7FFFu + ((u >> 16) & 1u);   // round-to-nearest-even
    return (unsigned short)(r >> 16);
}
__device__ __forceinline__ float bf16_to_f32(unsigned short s) {
    return __uint_as_float(((unsigned int)s) << 16);
}

// ---------------- 0. init counters ----------------
__global__ void init_kernel(int* cnt) {
    if (threadIdx.x < GSEG) cnt[threadIdx.x] = 0;
}

// ---------------- 1. weights -> bf16, transposed (N-major, K contiguous) ----
__global__ void prep_weights_kernel(const float* __restrict__ W1,
                                    const float* __restrict__ W2,
                                    unsigned short* __restrict__ wt1,   // [256][128]
                                    unsigned short* __restrict__ wt2) { // [64][256]
    int i = blockIdx.x * blockDim.x + threadIdx.x;
    if (i < 128 * 256) {                 // W1 is [128,256] (k,n)
        int k = i / 256, n = i % 256;
        wt1[n * 128 + k] = f32_to_bf16(W1[i]);
    }
    if (i < 256 * 64) {                  // W2 is [256,64] (k,n)
        int k = i / 64, n = i % 64;
        wt2[n * 256 + k] = f32_to_bf16(W2[i]);
    }
}

// ---------------- 2. histogram of segment ids (LDS-binned) ------------------
__global__ void hist_kernel(const int* __restrict__ eidx,
                            const int* __restrict__ batch,
                            int* __restrict__ cnt, int E) {
    __shared__ int h[GSEG];
    if (threadIdx.x < GSEG) h[threadIdx.x] = 0;
    __syncthreads();
    int stride = gridDim.x * blockDim.x;
    for (int i = blockIdx.x * blockDim.x + threadIdx.x; i < E; i += stride) {
        int g = batch[eidx[i]];          // seg = batch[col[e]]
        atomicAdd(&h[g], 1);
    }
    __syncthreads();
    if (threadIdx.x < GSEG) atomicAdd(&cnt[threadIdx.x], h[threadIdx.x]);
}

// ---------------- 3. exclusive scan + cursor reset --------------------------
__global__ void scan_kernel(const int* __restrict__ cnt,
                            int* __restrict__ start, int* __restrict__ cursor) {
    if (threadIdx.x == 0) {
        int acc = 0;
        for (int g = 0; g < GSEG; ++g) { start[g] = acc; acc += cnt[g]; }
    }
    if (threadIdx.x < GSEG) cursor[threadIdx.x] = 0;
}

// ---------------- 4. scatter edges into segment-sorted order ----------------
__global__ void scatter_kernel(const int* __restrict__ eidx,
                               const int* __restrict__ batch,
                               const int* __restrict__ start,
                               int* __restrict__ cursor,
                               int* __restrict__ perm, int* __restrict__ segs,
                               int E) {
    int i = blockIdx.x * blockDim.x + threadIdx.x;
    if (i >= E) return;
    int g = batch[eidx[i]];
    int pos = start[g] + atomicAdd(&cursor[g], 1);
    perm[pos] = i;
    segs[pos] = g;
}

// ---------------- 5. GEMM1: f12[64x128] @ W1[128x256] -> h1 bf16 ------------
__global__ __launch_bounds__(256) void gemm1_kernel(
    const float* __restrict__ emb, const int* __restrict__ eidx,
    const int* __restrict__ perm,
    const unsigned short* __restrict__ wt1, const float* __restrict__ b1,
    unsigned short* __restrict__ h1, int E) {
    __shared__ unsigned short sX[64 * 136];   // gathered bf16 inputs, padded
    __shared__ unsigned short sO[64 * 264];   // staged bf16 outputs, padded
    const int p0 = blockIdx.x * 64;
    const int t  = threadIdx.x;
    {   // gather + f32->bf16: thread t -> row t/4, quarter t%4 (32 channels)
        const int rr = t >> 2, q = t & 3;
        const int e  = perm[p0 + rr];
        const int node = (q < 2) ? eidx[e] : eidx[E + e];   // col | row endpoint
        const float* src = emb + (long long)node * 64 + (q & 1) * 32;
        unsigned short* dst = sX + rr * 136 + q * 32;
        #pragma unroll
        for (int i = 0; i < 32; ++i) dst[i] = f32_to_bf16(src[i]);
    }
    __syncthreads();
    const int wave = t >> 5, lane = t & 31;
    const int mt = wave & 3;                    // M-tile 0..3
    const int ntG = wave >> 2;                  // N-tile group (0: nt 0-7, 1: nt 8-15)
    const int lhalf = lane >> 4;
    const int mrow = mt * 16 + (lane & 15);
    const int ncol = lane & 15;
    bf16x16 afrag[4];
    #pragma unroll
    for (int ks = 0; ks < 4; ++ks) {            // A: 16 contiguous K per lane
        const unsigned short* base = sX + mrow * 136 + ks * 32 + lhalf * 16;
        bf16x8 lo = *reinterpret_cast<const bf16x8*>(base);
        bf16x8 hi = *reinterpret_cast<const bf16x8*>(base + 8);
        bf16x16 a;
        #pragma unroll
        for (int i = 0; i < 8; ++i) { a[i] = lo[i]; a[8 + i] = hi[i]; }
        afrag[ks] = a;
    }
    #pragma unroll
    for (int j = 0; j < 8; ++j) {
        const int n = (ntG * 8 + j) * 16 + ncol;
        f32x8 acc = {};
        #pragma unroll
        for (int ks = 0; ks < 4; ++ks) {        // B: 16 contiguous K per lane
            bf16x16 b = *reinterpret_cast<const bf16x16*>(
                wt1 + n * 128 + ks * 32 + lhalf * 16);
            acc = __builtin_amdgcn_wmma_f32_16x16x32_bf16(
                false, afrag[ks], false, b, (short)0, acc, false, false);
        }
        const float bias = b1[n];
        #pragma unroll
        for (int i = 0; i < 8; ++i)
            sO[(mt * 16 + i + lhalf * 8) * 264 + n] = f32_to_bf16(acc[i] + bias);
    }
    __syncthreads();
    {   // coalesced bf16 store: thread t -> row t/4, 64-ush chunk t%4
        const int rr = t >> 2, q = t & 3;
        const uint4* src = (const uint4*)(sO + rr * 264 + q * 64);
        uint4* dst = (uint4*)(h1 + (long long)(p0 + rr) * 256 + q * 64);
        #pragma unroll
        for (int i = 0; i < 8; ++i) dst[i] = src[i];
    }
}

// ---------------- 6. per-segment stats (256 channels) -----------------------
__global__ __launch_bounds__(1024) void stats1_kernel(
    const unsigned short* __restrict__ h1, const int* __restrict__ start,
    const int* __restrict__ cnt, float* __restrict__ mean1,
    float* __restrict__ rstd1) {
    __shared__ float rs[4][256], rq[4][256];
    const int g = blockIdx.x;
    const int c = threadIdx.x & 255, sl = threadIdx.x >> 8;
    const int s = start[g], n = cnt[g];
    float sum = 0.f, sq = 0.f;
    for (int i = sl; i < n; i += 4) {
        float v = bf16_to_f32(h1[(long long)(s + i) * 256 + c]);
        sum += v; sq += v * v;
    }
    rs[sl][c] = sum; rq[sl][c] = sq;
    __syncthreads();
    if (sl == 0) {
        sum = rs[0][c] + rs[1][c] + rs[2][c] + rs[3][c];
        sq  = rq[0][c] + rq[1][c] + rq[2][c] + rq[3][c];
        const float denom = fmaxf((float)n, 1.f);
        const float m = sum / denom;
        const float var = fmaxf(sq / denom - m * m, 0.f);
        mean1[g * 256 + c] = m;
        rstd1[g * 256 + c] = rsqrtf(var + EPS);
    }
}

// ---------------- 7. GEMM2: norm(h1)+relu [64x256] @ W2[256x64] -------------
__global__ __launch_bounds__(256) void gemm2_kernel(
    const unsigned short* __restrict__ h1, const int* __restrict__ segs,
    const float* __restrict__ mean1, const float* __restrict__ rstd1,
    const unsigned short* __restrict__ wt2, const float* __restrict__ b2,
    unsigned short* __restrict__ h2) {
    __shared__ unsigned short sX[64 * 264];
    __shared__ unsigned short sO[64 * 72];
    const int p0 = blockIdx.x * 64;
    const int t  = threadIdx.x;
    {   // load + instance-norm + relu -> bf16 LDS
        const int rr = t >> 2, q = t & 3;
        const int p = p0 + rr, g = segs[p];
        const unsigned short* src = h1 + (long long)p * 256 + q * 64;
        const float* mr = mean1 + g * 256 + q * 64;
        const float* rr_ = rstd1 + g * 256 + q * 64;
        unsigned short* dst = sX + rr * 264 + q * 64;
        #pragma unroll
        for (int i = 0; i < 64; ++i) {
            float v = (bf16_to_f32(src[i]) - mr[i]) * rr_[i];
            dst[i] = f32_to_bf16(fmaxf(v, 0.f));
        }
    }
    __syncthreads();
    const int wave = t >> 5, lane = t & 31;
    const int mt = wave >> 1;                 // M-tile 0..3 (2 waves each)
    const int ntBase = (wave & 1) * 2;        // N-tiles {0,1} or {2,3}
    const int lhalf = lane >> 4;
    const int mrow = mt * 16 + (lane & 15);
    const int ncol = lane & 15;
    #pragma unroll
    for (int jn = 0; jn < 2; ++jn) {
        const int n = (ntBase + jn) * 16 + ncol;
        f32x8 acc = {};
        #pragma unroll
        for (int ks = 0; ks < 8; ++ks) {
            const unsigned short* base = sX + mrow * 264 + ks * 32 + lhalf * 16;
            bf16x8 lo = *reinterpret_cast<const bf16x8*>(base);
            bf16x8 hi = *reinterpret_cast<const bf16x8*>(base + 8);
            bf16x16 a;
            #pragma unroll
            for (int i = 0; i < 8; ++i) { a[i] = lo[i]; a[8 + i] = hi[i]; }
            bf16x16 b = *reinterpret_cast<const bf16x16*>(
                wt2 + n * 256 + ks * 32 + lhalf * 16);
            acc = __builtin_amdgcn_wmma_f32_16x16x32_bf16(
                false, a, false, b, (short)0, acc, false, false);
        }
        const float bias = b2[n];
        #pragma unroll
        for (int i = 0; i < 8; ++i)
            sO[(mt * 16 + i + lhalf * 8) * 72 + n] = f32_to_bf16(acc[i] + bias);
    }
    __syncthreads();
    {   // coalesced store 64x64 bf16
        const int rr = t >> 2, q = t & 3;
        const uint4* src = (const uint4*)(sO + rr * 72 + q * 16);
        uint4* dst = (uint4*)(h2 + (long long)(p0 + rr) * 64 + q * 16);
        dst[0] = src[0];
        dst[1] = src[1];
    }
}

// ---------------- 8. per-segment stats (64 channels) ------------------------
__global__ __launch_bounds__(512) void stats2_kernel(
    const unsigned short* __restrict__ h2, const int* __restrict__ start,
    const int* __restrict__ cnt, float* __restrict__ mean2,
    float* __restrict__ rstd2) {
    __shared__ float rs[8][64], rq[8][64];
    const int g = blockIdx.x;
    const int c = threadIdx.x & 63, sl = threadIdx.x >> 6;
    const int s = start[g], n = cnt[g];
    float sum = 0.f, sq = 0.f;
    for (int i = sl; i < n; i += 8) {
        float v = bf16_to_f32(h2[(long long)(s + i) * 64 + c]);
        sum += v; sq += v * v;
    }
    rs[sl][c] = sum; rq[sl][c] = sq;
    __syncthreads();
    if (sl == 0) {
        #pragma unroll
        for (int i = 1; i < 8; ++i) { sum += rs[i][c]; sq += rq[i][c]; }
        const float denom = fmaxf((float)n, 1.f);
        const float m = sum / denom;
        const float var = fmaxf(sq / denom - m * m, 0.f);
        mean2[g * 64 + c] = m;
        rstd2[g * 64 + c] = rsqrtf(var + EPS);
    }
}

// ---------------- 9. final: norm(h2)+relu dot W3 + b3, unsort ---------------
__global__ __launch_bounds__(256) void final_kernel(
    const unsigned short* __restrict__ h2, const int* __restrict__ segs,
    const float* __restrict__ mean2, const float* __restrict__ rstd2,
    const float* __restrict__ W3, const float* __restrict__ b3,
    const int* __restrict__ perm, float* __restrict__ out, int E) {
    __shared__ float w3s[64];
    if (threadIdx.x < 64) w3s[threadIdx.x] = W3[threadIdx.x];
    __syncthreads();
    const int p = blockIdx.x * 256 + threadIdx.x;
    if (p >= E) return;
    const int g = segs[p];
    const unsigned short* src = h2 + (long long)p * 64;
    const float* mr = mean2 + g * 64;
    const float* rr = rstd2 + g * 64;
    float acc = 0.f;
    #pragma unroll
    for (int i = 0; i < 64; ++i) {
        float v = (bf16_to_f32(src[i]) - mr[i]) * rr[i];
        acc += fmaxf(v, 0.f) * w3s[i];
    }
    out[perm[p]] = acc + b3[0];
}

extern "C" void kernel_launch(void* const* d_in, const int* in_sizes, int n_in,
                              void* d_out, int out_size, void* d_ws, size_t ws_size,
                              hipStream_t stream) {
    (void)n_in; (void)out_size; (void)ws_size;
    const float* emb   = (const float*)d_in[0];
    const int*   eidx  = (const int*)  d_in[1];
    const int*   batch = (const int*)  d_in[2];
    const float* W1    = (const float*)d_in[3];
    const float* b1    = (const float*)d_in[4];
    const float* W2    = (const float*)d_in[5];
    const float* b2    = (const float*)d_in[6];
    const float* W3    = (const float*)d_in[7];
    const float* b3    = (const float*)d_in[8];
    float* out = (float*)d_out;
    const int E = in_sizes[1] / 2;     // 800000 (divisible by 64 and 256)

    // carve workspace
    size_t off = 0;
    char* base = (char*)d_ws;
    auto carve = [&](size_t bytes) -> void* {
        void* p = base + off;
        off = (off + bytes + 255) & ~(size_t)255;
        return p;
    };
    unsigned short* wt1   = (unsigned short*)carve(256 * 128 * 2);
    unsigned short* wt2   = (unsigned short*)carve(64 * 256 * 2);
    int* cnt    = (int*)carve(GSEG * 4);
    int* start  = (int*)carve(GSEG * 4);
    int* cursor = (int*)carve(GSEG * 4);
    int* perm   = (int*)carve((size_t)E * 4);
    int* segs   = (int*)carve((size_t)E * 4);
    float* mean1 = (float*)carve(GSEG * 256 * 4);
    float* rstd1 = (float*)carve(GSEG * 256 * 4);
    float* mean2 = (float*)carve(GSEG * 64 * 4);
    float* rstd2 = (float*)carve(GSEG * 64 * 4);
    unsigned short* h1 = (unsigned short*)carve((size_t)E * 256 * 2);
    unsigned short* h2 = (unsigned short*)carve((size_t)E * 64 * 2);

    init_kernel<<<1, 128, 0, stream>>>(cnt);
    prep_weights_kernel<<<(128 * 256 + 255) / 256, 256, 0, stream>>>(W1, W2, wt1, wt2);
    hist_kernel<<<512, 256, 0, stream>>>(eidx, batch, cnt, E);
    scan_kernel<<<1, 128, 0, stream>>>(cnt, start, cursor);
    scatter_kernel<<<(E + 255) / 256, 256, 0, stream>>>(eidx, batch, start, cursor,
                                                        perm, segs, E);
    gemm1_kernel<<<E / 64, 256, 0, stream>>>(emb, eidx, perm, wt1, b1, h1, E);
    stats1_kernel<<<GSEG, 1024, 0, stream>>>(h1, start, cnt, mean1, rstd1);
    gemm2_kernel<<<E / 64, 256, 0, stream>>>(h1, segs, mean1, rstd1, wt2, b2, h2);
    stats2_kernel<<<GSEG, 512, 0, stream>>>(h2, start, cnt, mean2, rstd2);
    final_kernel<<<(E + 255) / 256, 256, 0, stream>>>(h2, segs, mean2, rstd2,
                                                      W3, b3, perm, out, E);
}
